// TGN_84078279786708
// MI455X (gfx1250) — matrix operand order repaired and checked by reading, not compiled, not run on hardware
//
#include <hip/hip_runtime.h>
#include <hip/hip_bf16.h>

typedef __attribute__((ext_vector_type(16))) _Float16 v16h;
typedef __attribute__((ext_vector_type(8)))  _Float16 v8h;
typedef __attribute__((ext_vector_type(4)))  _Float16 v4h;
typedef __attribute__((ext_vector_type(8)))  float    v8f;

#define E_TOTAL 262144
#define KDIM    512      // concat width: 4 * 128
#define HID     256
#define M_BLK   128      // edges per workgroup (8 m-tiles of 16)
#define ROWH    520      // LDS row stride in halves (512 + 8 pad -> bank-conflict-free b128)

// ---- prolog: W1 fp32 [256][512] -> f16 in workspace ------------------------
__global__ void w1_to_f16(const float* __restrict__ W1, _Float16* __restrict__ W1h) {
    int i = (blockIdx.x * blockDim.x + threadIdx.x) * 4;   // 131072 elems / 4
    float4 v = *(const float4*)(W1 + i);
    v4h h = { (_Float16)v.x, (_Float16)v.y, (_Float16)v.z, (_Float16)v.w };
    *(v4h*)(W1h + i) = h;
}

// ---- main: gather -> LDS, WMMA GEMM (4m x 4n tiles/wave), fused epilogue ---
__global__ __launch_bounds__(256)
void tgn_scorer(const int* __restrict__ src, const int* __restrict__ dst,
                const float* __restrict__ node_emb, const float* __restrict__ mem_tab,
                const _Float16* __restrict__ W1h, const float* __restrict__ b1,
                const float* __restrict__ W2, const float* __restrict__ b2,
                float* __restrict__ out) {
    __shared__ _Float16 sA[M_BLK * ROWH];   // 133120 B activation tile (f16)
    __shared__ float    sPart[4][M_BLK];    // per-n-quarter partial dots

    const int tid   = threadIdx.x;
    const int ebase = blockIdx.x * M_BLK;

    // -- cooperative gather + fp32->f16 convert into LDS --------------------
    // concat layout per edge row: [emb[src] | emb[dst] | mem[src] | mem[dst]]
    for (int it = tid; it < M_BLK * 128; it += 256) {
        int e   = it >> 7;        // edge within block
        int c4  = it & 127;       // float4 column (0..127) -> half col = c4*4
        int tbl = c4 >> 5;        // 0..3 selects table/index
        int idx = (tbl & 1) ? dst[ebase + e] : src[ebase + e];
        const float* bp = ((tbl < 2) ? node_emb : mem_tab)
                          + (size_t)idx * 128 + (c4 & 31) * 4;
        float4 v = *(const float4*)bp;
        v4h h = { (_Float16)v.x, (_Float16)v.y, (_Float16)v.z, (_Float16)v.w };
        *(v4h*)(&sA[e * ROWH + c4 * 4]) = h;
    }
    __syncthreads();

    // -- per-wave tile assignment: 4 m-tiles x 4 n-tiles per wave ------------
    const int lane = tid & 31;
    const int wave = tid >> 5;      // 0..7
    const int mq   = wave & 1;      // m-tiles [4*mq .. 4*mq+3]  (rows mq*64..)
    const int nq   = wave >> 1;     // n-tiles [4*nq .. 4*nq+3]  (cols nq*64..)
    const int lrow = lane & 15;     // A: M index / B,C: N index
    const int half = lane >> 4;     // A: K-half / B: K-group / C: M-half

    const _Float16* aRow[4];
    #pragma unroll
    for (int j = 0; j < 4; ++j)
        aRow[j] = &sA[(mq * 64 + j * 16 + lrow) * ROWH];

    float w2v[4], b1v[4];
    #pragma unroll
    for (int t = 0; t < 4; ++t) {
        int n = nq * 64 + t * 16 + lrow;
        w2v[t] = W2[n];
        b1v[t] = b1[n];
    }

    const v8f zero8 = {0.f,0.f,0.f,0.f,0.f,0.f,0.f,0.f};
    v8f acc[4][4];                  // [m-tile j][n-tile t]
    #pragma unroll
    for (int j = 0; j < 4; ++j)
        #pragma unroll
        for (int t = 0; t < 4; ++t) acc[j][t] = zero8;

    // -- K loop: 16 steps; 8 fragment loads feed 16 WMMAs each step ---------
    for (int ks = 0; ks < 16; ++ks) {
        const int kb = ks * 32;

        // A fragments (ISA 16-bit A 16x32 layout): two b128 LDS loads each
        v16h a[4];
        #pragma unroll
        for (int j = 0; j < 4; ++j) {
            v8h lo = *(const v8h*)(aRow[j] + kb + half * 8);
            v8h hi = *(const v8h*)(aRow[j] + kb + 16 + half * 8);
            #pragma unroll
            for (int i = 0; i < 8; ++i) { a[j][i] = lo[i]; a[j][i + 8] = hi[i]; }
        }

        // B fragments: lane holds K [8*half..+8) and [16+8*half..+8) of row n = nq*64+t*16+lrow
        const _Float16* bp = W1h + (size_t)(nq * 64 + lrow) * KDIM
                                 + kb + half * 8;
        v16h b[4];
        #pragma unroll
        for (int t = 0; t < 4; ++t) {
            v8h lo = *(const v8h*)(bp + t * 16 * KDIM);
            v8h hi = *(const v8h*)(bp + t * 16 * KDIM + 16);
            #pragma unroll
            for (int i = 0; i < 8; ++i) { b[t][i] = lo[i]; b[t][i + 8] = hi[i]; }
        }

        #pragma unroll
        for (int t = 0; t < 4; ++t)
            #pragma unroll
            for (int j = 0; j < 4; ++j)
                acc[j][t] = __builtin_amdgcn_wmma_f32_16x16x32_f16(
                               false, a[j], false, b[t], (short)0,
                               acc[j][t], false, false);
    }

    // -- fused epilogue: relu(h1 + b1) . W2, reduce over N ------------------
    #pragma unroll
    for (int j = 0; j < 4; ++j) {
        float p[8];
        #pragma unroll
        for (int r = 0; r < 8; ++r) p[r] = 0.f;
        #pragma unroll
        for (int t = 0; t < 4; ++t) {
            float bb = b1v[t], w = w2v[t];
            #pragma unroll
            for (int r = 0; r < 8; ++r)
                p[r] += fmaxf(acc[j][t][r] + bb, 0.f) * w;
        }
        float outv = 0.f;
        #pragma unroll
        for (int r = 0; r < 8; ++r) {
            float s = p[r];
            s += __shfl_xor(s, 1);
            s += __shfl_xor(s, 2);
            s += __shfl_xor(s, 4);
            s += __shfl_xor(s, 8);     // all 16 lanes of the half-group hold sum
            if (lrow == r) outv = s;   // lane r keeps row r+8*half
        }
        if (lrow < 8)
            sPart[nq][mq * 64 + j * 16 + half * 8 + lrow] = outv;
    }
    __syncthreads();

    if (tid < M_BLK)
        out[ebase + tid] = sPart[0][tid] + sPart[1][tid]
                         + sPart[2][tid] + sPart[3][tid] + b2[0];
}

extern "C" void kernel_launch(void* const* d_in, const int* in_sizes, int n_in,
                              void* d_out, int out_size, void* d_ws, size_t ws_size,
                              hipStream_t stream) {
    const int*   src      = (const int*)  d_in[0];
    const int*   dst      = (const int*)  d_in[1];
    const float* node_emb = (const float*)d_in[4];
    const float* mem_tab  = (const float*)d_in[5];
    const float* W1       = (const float*)d_in[10];
    const float* b1       = (const float*)d_in[11];
    const float* W2       = (const float*)d_in[12];
    const float* b2       = (const float*)d_in[13];
    float*    out = (float*)d_out;
    _Float16* W1h = (_Float16*)d_ws;          // 256 KB scratch

    w1_to_f16<<<(HID * KDIM / 4) / 256, 256, 0, stream>>>(W1, W1h);
    tgn_scorer<<<E_TOTAL / M_BLK, 256, 0, stream>>>(
        src, dst, node_emb, mem_tab, W1h, b1, W2, b2, out);
}
